// modified_Net0_73735998538050
// MI455X (gfx1250) — matrix operand order repaired and checked by reading, not compiled, not run on hardware
//
#include <hip/hip_runtime.h>
#include <math.h>

typedef __attribute__((ext_vector_type(16))) __bf16 v16bf;
typedef __attribute__((ext_vector_type(8)))  float  v8f;
typedef __attribute__((ext_vector_type(4)))  unsigned int v4u;

#define GF_RELU   1
#define GF_TRANSA 2
#define GF_TRANSB 4

// ---------------------------------------------------------------------------
// Stage one 16-element K-segment of a tile row into LDS as bf16.
// KCONTIG = K index is contiguous in memory (element = G[row*ld + k]);
// otherwise strided (element = G[k*ld + row]).
// Fast path: full segment in-bounds -> vector loads + 2x ds_store_b128.
// Cold path (K edge only, e.g. K=3 layer): compact scalar loop.
// ---------------------------------------------------------------------------
template <bool KCONTIG>
__device__ __forceinline__ void stage_seg(const float* __restrict__ G, int ld,
                                          int rows, int K, int grow,
                                          int k0, int kh, __bf16* __restrict__ dst)
{
    const int kb = k0 + kh;
    if (grow < rows && kb + 16 <= K) {
        float v[16];
        if (KCONTIG) {
            const float4* g = (const float4*)&G[(size_t)grow * ld + kb];
            float4 a = g[0], b = g[1], c = g[2], d = g[3];
            v[0]=a.x;  v[1]=a.y;  v[2]=a.z;  v[3]=a.w;
            v[4]=b.x;  v[5]=b.y;  v[6]=b.z;  v[7]=b.w;
            v[8]=c.x;  v[9]=c.y;  v[10]=c.z; v[11]=c.w;
            v[12]=d.x; v[13]=d.y; v[14]=d.z; v[15]=d.w;
        } else {
#pragma unroll
            for (int j = 0; j < 16; ++j)
                v[j] = G[(size_t)(kb + j) * ld + grow];
        }
        union { __bf16 h[16]; v4u u[2]; } cv;
#pragma unroll
        for (int j = 0; j < 16; ++j) cv.h[j] = (__bf16)v[j];
        ((v4u*)dst)[0] = cv.u[0];
        ((v4u*)dst)[1] = cv.u[1];
    } else {
#pragma unroll 1
        for (int j = 0; j < 16; ++j) {
            const int k = kb + j;
            float x = 0.f;
            if (k < K && grow < rows)
                x = KCONTIG ? G[(size_t)grow * ld + k] : G[(size_t)k * ld + grow];
            dst[j] = (__bf16)x;
        }
    }
}

// ---------------------------------------------------------------------------
// Generic batched GEMM: C(M,Nn) = op(A)(M,K) @ op(B)(K,Nn) [+bias][+Cadd][relu]
// bf16 WMMA 16x16x32, fp32 accumulate.
// Block = 256 thr (8 waves), tile 128x128; wave = 32x64 (2x4 WMMA tiles).
// ---------------------------------------------------------------------------
template <int FLAGS>
__global__ __launch_bounds__(256) void gemm_bf16_wmma(
    const float* __restrict__ A, const float* __restrict__ Bm,
    const float* __restrict__ bias, const float* __restrict__ Cadd,
    float* __restrict__ C,
    int M, int Nn, int K,
    int ldA, int ldB, int ldAdd, int ldC,
    long sA, long sB, long sAdd, long sC)
{
    constexpr bool TA   = (FLAGS & GF_TRANSA) != 0;
    constexpr bool TB   = (FLAGS & GF_TRANSB) != 0;
    constexpr bool RELU = (FLAGS & GF_RELU) != 0;

    __shared__ __bf16 As[128][40];   // 128 rows (M) x 32 k, padded to 40
    __shared__ __bf16 Bs[128][40];   // 128 rows (N) x 32 k, padded to 40

    const int b = blockIdx.z;
    A += (size_t)b * sA;
    Bm += (size_t)b * sB;
    C += (size_t)b * sC;
    if (Cadd) Cadd += (size_t)b * sAdd;

    const int m0 = blockIdx.y * 128, n0 = blockIdx.x * 128;
    const int tid = threadIdx.x;
    const int lane = tid & 31, wid = tid >> 5;
    const int wm = (wid >> 1) * 32;       // 4 waves along M
    const int wn = (wid & 1) * 64;        // 2 waves along N
    const int half = lane >> 4, l15 = lane & 15;

    const int ldr = tid >> 1;             // 0..127 : LDS row this thread fills
    const int kh  = (tid & 1) * 16;       // 0 or 16

    v8f acc[2][4] = {};

    for (int k0 = 0; k0 < K; k0 += 32) {
        stage_seg<!TA>(A, ldA, M, K, m0 + ldr, k0, kh, &As[ldr][kh]);
        stage_seg<TB>(Bm, ldB, Nn, K, n0 + ldr, k0, kh, &Bs[ldr][kh]);
        __syncthreads();

        union Frag { v16bf bf; v4u u[2]; };
        Frag fa[2], fb[4];
#pragma unroll
        for (int t = 0; t < 2; ++t) {
            const int row = wm + t * 16 + l15;
            // A 16x32 bf16 layout: VGPR0-3 = K half*8..+7, VGPR4-7 = 16+half*8..+7
            fa[t].u[0] = *(const v4u*)&As[row][half * 8];
            fa[t].u[1] = *(const v4u*)&As[row][16 + half * 8];
        }
#pragma unroll
        for (int t = 0; t < 4; ++t) {
            const int col = wn + t * 16 + l15;
            // B 32x16 bf16 layout: VGPR0-3 = K half*8..+7, VGPR4-7 = 16+half*8..+7
            fb[t].u[0] = *(const v4u*)&Bs[col][half * 8];
            fb[t].u[1] = *(const v4u*)&Bs[col][16 + half * 8];
        }
#pragma unroll
        for (int ti = 0; ti < 2; ++ti)
#pragma unroll
            for (int tj = 0; tj < 4; ++tj)
                acc[ti][tj] = __builtin_amdgcn_wmma_f32_16x16x32_bf16(
                    false, fa[ti].bf, false, fb[tj].bf,
                    (short)0, acc[ti][tj], false, false);
        __syncthreads();
    }

    // ---- epilogue: C layout 16x16 f32 (VGPR r: M = r + 8*half, N = l15) ----
#pragma unroll
    for (int ti = 0; ti < 2; ++ti) {
#pragma unroll
        for (int tj = 0; tj < 4; ++tj) {
            const int col = n0 + wn + tj * 16 + l15;
#pragma unroll
            for (int r = 0; r < 8; ++r) {
                const int row = m0 + wm + ti * 16 + half * 8 + r;
                if (row < M && col < Nn) {
                    float v = acc[ti][tj][r];
                    if (bias) v += bias[col];
                    if (Cadd) v += Cadd[(size_t)row * ldAdd + col];
                    if (RELU) v = v > 0.f ? v : 0.f;
                    C[(size_t)row * ldC + col] = v;
                }
            }
        }
    }
}

// ---------------------------------------------------------------------------
// Row softmax (in place) with input scale. One block per (row, batch).
// ---------------------------------------------------------------------------
__global__ void softmax_rows(float* X, int cols, float scale)
{
    const int row = blockIdx.x, b = blockIdx.y, rowsPerB = gridDim.x;
    float* p = X + ((size_t)b * rowsPerB + row) * cols;
    __shared__ float sh[256];
    const int t = threadIdx.x;

    float m = -3.4e38f;
    for (int c = t; c < cols; c += 256) m = fmaxf(m, p[c] * scale);
    sh[t] = m; __syncthreads();
    for (int s = 128; s > 0; s >>= 1) { if (t < s) sh[t] = fmaxf(sh[t], sh[t + s]); __syncthreads(); }
    m = sh[0]; __syncthreads();

    float sum = 0.f;
    for (int c = t; c < cols; c += 256) sum += __expf(p[c] * scale - m);
    sh[t] = sum; __syncthreads();
    for (int s = 128; s > 0; s >>= 1) { if (t < s) sh[t] += sh[t + s]; __syncthreads(); }
    const float inv = 1.f / sh[0];

    for (int c = t; c < cols; c += 256) p[c] = __expf(p[c] * scale - m) * inv;
}

// ---------------------------------------------------------------------------
// Sinkhorn (log domain) on padded (N+1)x(N+1) matrices
// ---------------------------------------------------------------------------
__global__ void init_la(const float* __restrict__ S, float* __restrict__ la, int N)
{
    const int Np1 = N + 1;
    const int b = blockIdx.y;
    const int id = blockIdx.x * 256 + threadIdx.x;
    if (id >= Np1 * Np1) return;
    const int r = id / Np1, c = id % Np1;
    la[(size_t)b * Np1 * Np1 + id] =
        (r < N && c < N) ? S[(size_t)b * N * N + (size_t)r * N + c] : 0.f;
}

// rows 0..N-1: subtract logsumexp over all N+1 columns. Block per (row,b).
__global__ void sink_row(float* la, int N)
{
    const int Np1 = N + 1;
    const int b = blockIdx.y, r = blockIdx.x;
    float* p = la + (size_t)b * Np1 * Np1 + (size_t)r * Np1;
    __shared__ float sh[256];
    const int t = threadIdx.x;

    float m = -3.4e38f;
    for (int c = t; c < Np1; c += 256) m = fmaxf(m, p[c]);
    sh[t] = m; __syncthreads();
    for (int s = 128; s > 0; s >>= 1) { if (t < s) sh[t] = fmaxf(sh[t], sh[t + s]); __syncthreads(); }
    m = sh[0]; __syncthreads();

    float sum = 0.f;
    for (int c = t; c < Np1; c += 256) sum += __expf(p[c] - m);
    sh[t] = sum; __syncthreads();
    for (int s = 128; s > 0; s >>= 1) { if (t < s) sh[t] += sh[t + s]; __syncthreads(); }
    const float lse = m + __logf(sh[0]);

    for (int c = t; c < Np1; c += 256) p[c] -= lse;
}

// cols 0..N-1: subtract logsumexp over all N+1 rows. Thread per column.
__global__ void sink_col(float* la, int N)
{
    const int Np1 = N + 1;
    const int b = blockIdx.y;
    const int c = blockIdx.x * blockDim.x + threadIdx.x;
    if (c >= N) return;
    float* p = la + (size_t)b * Np1 * Np1 + c;

    float m = -3.4e38f;
    for (int r = 0; r <= N; ++r) m = fmaxf(m, p[(size_t)r * Np1]);
    float sum = 0.f;
    for (int r = 0; r <= N; ++r) sum += __expf(p[(size_t)r * Np1] - m);
    const float lse = m + __logf(sum);
    for (int r = 0; r <= N; ++r) p[(size_t)r * Np1] -= lse;
}

__global__ void exp_interior(const float* __restrict__ la, float* __restrict__ S, int N)
{
    const int Np1 = N + 1;
    const int b = blockIdx.y;
    const int id = blockIdx.x * 256 + threadIdx.x;
    if (id >= N * N) return;
    const int r = id / N, c = id % N;
    S[(size_t)b * N * N + id] = __expf(la[(size_t)b * Np1 * Np1 + (size_t)r * Np1 + c]);
}

// ---------------------------------------------------------------------------
// Misc small kernels
// ---------------------------------------------------------------------------
__global__ void symmetrize(const float* __restrict__ Aff, float* __restrict__ Msym, int d)
{
    const int id = blockIdx.x * 256 + threadIdx.x;
    if (id >= d * d) return;
    const int r = id / d, c = id % d;
    Msym[id] = 0.5f * (Aff[id] + Aff[(size_t)c * d + r]);
}

// per-feature-column softmax stats over nodes (axis=1 softmax)
__global__ void colstats(const float* __restrict__ E, int n, int d,
                         float* __restrict__ cmax, float* __restrict__ csum)
{
    const int b = blockIdx.y;
    const int f = blockIdx.x * blockDim.x + threadIdx.x;
    if (f >= d) return;
    const float* p = E + (size_t)b * n * d + f;
    float m = -3.4e38f;
    for (int r = 0; r < n; ++r) m = fmaxf(m, p[(size_t)r * d]);
    float s = 0.f;
    for (int r = 0; r < n; ++r) s += __expf(p[(size_t)r * d] - m);
    cmax[(size_t)b * d + f] = m;
    csum[(size_t)b * d + f] = s;
}

__global__ void score_rows(const float* __restrict__ E, const float* __restrict__ cmax,
                           const float* __restrict__ csum, float* __restrict__ score, int d)
{
    const int b = blockIdx.y, r = blockIdx.x, n = gridDim.x;
    const float* p = E + ((size_t)b * n + r) * d;
    __shared__ float sh[256];
    const int t = threadIdx.x;
    float s = 0.f;
    for (int f = t; f < d; f += 256)
        s += __expf(p[f] - cmax[(size_t)b * d + f]) / csum[(size_t)b * d + f];
    sh[t] = s; __syncthreads();
    for (int st = 128; st > 0; st >>= 1) { if (t < st) sh[t] += sh[t + st]; __syncthreads(); }
    if (t == 0) score[(size_t)b * n + r] = sh[0];
}

// full bitonic sort (descending, ties -> smaller index) then take first k
__global__ void topk_sort(const float* __restrict__ score, int* __restrict__ idx, int k)
{
    const int n = blockDim.x;   // power of two (1024 or 512)
    const int b = blockIdx.x;
    const int t = threadIdx.x;
    __shared__ float sv[1024];
    __shared__ int   si[1024];
    sv[t] = score[(size_t)b * n + t];
    si[t] = t;
    __syncthreads();
    for (int ksz = 2; ksz <= n; ksz <<= 1) {
        for (int j = ksz >> 1; j > 0; j >>= 1) {
            const int x = t ^ j;
            if (x > t) {
                const float a = sv[t], bb = sv[x];
                const int ia = si[t], ib = si[x];
                const bool t_gt = (a > bb) || (a == bb && ia < ib);
                const bool desc = ((t & ksz) == 0);
                if (desc ? !t_gt : t_gt) { sv[t] = bb; sv[x] = a; si[t] = ib; si[x] = ia; }
            }
            __syncthreads();
        }
    }
    if (t < k) idx[(size_t)b * k + t] = si[t];
}

__global__ void gather_rows(const float* __restrict__ E, const int* __restrict__ idx,
                            float* __restrict__ Xn, int n, int k, int d)
{
    const int b = blockIdx.y;
    const int id = blockIdx.x * 256 + threadIdx.x;
    if (id >= k * d) return;
    const int f = id % d, n2 = id / d;
    Xn[((size_t)b * k + n2) * d + f] =
        E[((size_t)b * n + idx[(size_t)b * k + n2]) * d + f];
}

// ---------------------------------------------------------------------------
// Host orchestration
// ---------------------------------------------------------------------------
extern "C" void kernel_launch(void* const* d_in, const int* in_sizes, int n_in,
                              void* d_out, int out_size, void* d_ws, size_t ws_size,
                              hipStream_t stream)
{
    (void)in_sizes; (void)n_in; (void)out_size; (void)ws_size;
    const int Bb = 4;
    const int DIMS[3] = {512, 256, 128};
    const int baseIdx[3] = {4, 15, 27};
    const int DINS[3] = {3, 512, 256};

    auto F = [&](int ix) { return (const float*)d_in[ix]; };
    const float* Psrc = F(0);
    const float* Ptgt = F(1);
    const float* Asrc = F(2);
    const float* Atgt = F(3);

    // bump allocator over workspace
    char* w = (char*)d_ws;
    auto allocf = [&](size_t nfl) {
        float* p = (float*)w;
        w += ((nfl * sizeof(float)) + 255) & ~(size_t)255;
        return p;
    };
    float* emb2d_s = allocf(4ull * 1024 * 1024);
    float* emb2d_t = allocf(4ull * 1024 * 1024);
    float* catC_s  = allocf(4ull * 1024 * 1024);
    float* catC_t  = allocf(4ull * 1024 * 1024);
    float* Sbuf    = allocf(4ull * 1024 * 1024);
    float* labuf   = allocf(4ull * 1025 * 1025);
    float* Ubuf    = allocf(4ull * 1024 * 512);
    float* Wbuf    = allocf(4ull * 1024 * 512);
    float* Tmp     = allocf(4ull * 1024 * 512);
    float* Tproj   = allocf(4ull * 512 * 256);
    float* Alog_s  = allocf(4ull * 512 * 512);
    float* Alog_t  = allocf(4ull * 512 * 512);
    float* Msym    = allocf(512ull * 512);
    float* Xn_s    = allocf(4ull * 512 * 512);
    float* Xn_t    = allocf(4ull * 512 * 512);
    float* cmax    = allocf(4ull * 512);
    float* csum    = allocf(4ull * 512);
    float* scoreb  = allocf(4ull * 1024);
    int*   idxb    = (int*)allocf(4ull * 1024);

    auto GEMM = [&](const float* Ap, const float* Bp, const float* biasp,
                    const float* Cp, float* Cd,
                    int M, int Nn, int K, int lda, int ldb, int ldadd, int ldc,
                    long sa, long sb, long sadd, long sc, int fl) {
        dim3 g((Nn + 127) / 128, (M + 127) / 128, Bb);
        switch (fl) {
        case 0:
            gemm_bf16_wmma<0><<<g, 256, 0, stream>>>(Ap, Bp, biasp, Cp, Cd, M, Nn, K,
                                                     lda, ldb, ldadd, ldc, sa, sb, sadd, sc);
            break;
        case GF_RELU:
            gemm_bf16_wmma<GF_RELU><<<g, 256, 0, stream>>>(Ap, Bp, biasp, Cp, Cd, M, Nn, K,
                                                           lda, ldb, ldadd, ldc, sa, sb, sadd, sc);
            break;
        case GF_TRANSA:
            gemm_bf16_wmma<GF_TRANSA><<<g, 256, 0, stream>>>(Ap, Bp, biasp, Cp, Cd, M, Nn, K,
                                                             lda, ldb, ldadd, ldc, sa, sb, sadd, sc);
            break;
        default:
            gemm_bf16_wmma<GF_TRANSB><<<g, 256, 0, stream>>>(Ap, Bp, biasp, Cp, Cd, M, Nn, K,
                                                             lda, ldb, ldadd, ldc, sa, sb, sadd, sc);
            break;
        }
    };

    const float* Xs = Psrc;
    const float* Xt = Ptgt;
    int Ncur = 1024;

    for (int i = 0; i < 3; ++i) {
        const int d = DIMS[i], din = DINS[i], d2 = 2 * d;
        const int o = baseIdx[i];
        const float* Wp = F(o + 0); const float* bp = F(o + 1);
        const float* We = F(o + 2); const float* be = F(o + 3);
        const float* Wt = (i > 0) ? F(o + 4) : nullptr;
        const int oo = (i > 0) ? o + 5 : o + 4;
        const float* Wa = F(oo + 0); const float* ba = F(oo + 1);
        const float* Wu = F(oo + 2); const float* bu = F(oo + 3);
        const float* Aff = F(oo + 4);
        const float* Wc = F(oo + 5); const float* bc = F(oo + 6);

        const long sX = (long)Ncur * din, s2d = (long)Ncur * d2;
        const long sd = (long)Ncur * d,   sNN = (long)Ncur * Ncur;

        // ---- 1. node / edge projections -> emb2d = [relu(X@Wp+bp), relu(X@We+be)]
        GEMM(Xs, Wp, bp, nullptr, emb2d_s,     Ncur, d, din, din, d, 0, d2, sX, 0, 0, s2d, GF_RELU);
        GEMM(Xs, We, be, nullptr, emb2d_s + d, Ncur, d, din, din, d, 0, d2, sX, 0, 0, s2d, GF_RELU);
        GEMM(Xt, Wp, bp, nullptr, emb2d_t,     Ncur, d, din, din, d, 0, d2, sX, 0, 0, s2d, GF_RELU);
        GEMM(Xt, We, be, nullptr, emb2d_t + d, Ncur, d, din, din, d, 0, d2, sX, 0, 0, s2d, GF_RELU);

        // ---- 2. attention matrices ----
        const float* AttS; const float* AttT; int ldAtt; long sAtt;
        if (i == 0) {
            AttS = Asrc; AttT = Atgt; ldAtt = Ncur; sAtt = sNN;
        } else {
            const float scale = 1.0f / sqrtf((float)d);
            GEMM(emb2d_s, Wt, nullptr, nullptr, Tproj, Ncur, d, d2, d2, d, 0, d, s2d, 0, 0, sd, 0);
            GEMM(Tproj, Tproj, nullptr, nullptr, Alog_s, Ncur, Ncur, d, d, d, 0, Ncur, sd, sd, 0, sNN, GF_TRANSB);
            softmax_rows<<<dim3(Ncur, Bb), 256, 0, stream>>>(Alog_s, Ncur, scale);
            GEMM(emb2d_t, Wt, nullptr, nullptr, Tproj, Ncur, d, d2, d2, d, 0, d, s2d, 0, 0, sd, 0);
            GEMM(Tproj, Tproj, nullptr, nullptr, Alog_t, Ncur, Ncur, d, d, d, 0, Ncur, sd, sd, 0, sNN, GF_TRANSB);
            softmax_rows<<<dim3(Ncur, Bb), 256, 0, stream>>>(Alog_t, Ncur, scale);
            AttS = Alog_s; AttT = Alog_t; ldAtt = Ncur; sAtt = sNN;
        }

        // ---- 3. GNN update: emb = relu(A@(emb@Wa+ba) + emb@Wu+bu) -> catC[:, 0:d)
        for (int side = 0; side < 2; ++side) {
            const float* emb = side ? emb2d_t : emb2d_s;
            const float* Att = side ? AttT : AttS;
            float* cat = side ? catC_t : catC_s;
            GEMM(emb, Wa, ba, nullptr, Ubuf, Ncur, d, d2, d2, d, 0, d, s2d, 0, 0, sd, 0);
            GEMM(emb, Wu, bu, nullptr, Wbuf, Ncur, d, d2, d2, d, 0, d, s2d, 0, 0, sd, 0);
            GEMM(Att, Ubuf, nullptr, Wbuf, cat, Ncur, d, Ncur, ldAtt, d, d, d2, sAtt, sd, sd, s2d, GF_RELU);
        }

        // ---- 4. affinity scores: Slog = emb_s @ sym(Aff) @ emb_t^T ----
        symmetrize<<<(d * d + 255) / 256, 256, 0, stream>>>(Aff, Msym, d);
        GEMM(catC_s, Msym, nullptr, nullptr, Tmp, Ncur, d, d, d2, d, 0, d, s2d, 0, 0, sd, 0);
        GEMM(Tmp, catC_t, nullptr, nullptr, Sbuf, Ncur, Ncur, d, d, d2, 0, Ncur, sd, s2d, 0, sNN, GF_TRANSB);

        // ---- 5. Sinkhorn (log domain, padded) + exp ----
        const int Np1 = Ncur + 1;
        init_la<<<dim3((Np1 * Np1 + 255) / 256, Bb), 256, 0, stream>>>(Sbuf, labuf, Ncur);
        for (int it = 0; it < 20; ++it) {
            sink_row<<<dim3(Ncur, Bb), 256, 0, stream>>>(labuf, Ncur);
            sink_col<<<dim3((Ncur + 255) / 256, Bb), 256, 0, stream>>>(labuf, Ncur);
        }
        float* Sdst = (i == 2) ? (float*)d_out : Sbuf;
        exp_interior<<<dim3((Ncur * Ncur + 255) / 256, Bb), 256, 0, stream>>>(labuf, Sdst, Ncur);
        if (i == 2) break;   // last layer returns s

        // ---- 6. cross-graph embedding + combine ----
        // catC_s[:, d:2d) = S @ emb_t ; catC_t[:, d:2d) = S^T @ emb_s
        GEMM(Sbuf, catC_t, nullptr, nullptr, catC_s + d, Ncur, d, Ncur, Ncur, d2, 0, d2, sNN, s2d, 0, s2d, 0);
        GEMM(Sbuf, catC_s, nullptr, nullptr, catC_t + d, Ncur, d, Ncur, Ncur, d2, 0, d2, sNN, s2d, 0, s2d, GF_TRANSA);
        GEMM(catC_s, Wc, bc, nullptr, Ubuf, Ncur, d, d2, d2, d, 0, d, s2d, 0, 0, sd, 0);  // emb1
        GEMM(catC_t, Wc, bc, nullptr, Wbuf, Ncur, d, d2, d2, d, 0, d, s2d, 0, 0, sd, 0);  // emb2

        // ---- 7. top-k node pooling ----
        const int k = DIMS[i];
        for (int side = 0; side < 2; ++side) {
            const float* E = side ? Wbuf : Ubuf;
            float* Xn = side ? Xn_t : Xn_s;
            colstats<<<dim3((d + 255) / 256, Bb), 256, 0, stream>>>(E, Ncur, d, cmax, csum);
            score_rows<<<dim3(Ncur, Bb), 256, 0, stream>>>(E, cmax, csum, scoreb, d);
            topk_sort<<<Bb, Ncur, 0, stream>>>(scoreb, idxb, k);
            gather_rows<<<dim3((k * d + 255) / 256, Bb), 256, 0, stream>>>(E, idxb, Xn, Ncur, k, d);
        }
        Xs = Xn_s; Xt = Xn_t; Ncur = k;
    }
}
